// DeepMD_1529008357941
// MI455X (gfx1250) — matrix operand order repaired and checked by reading, not compiled, not run on hardware
//
#include <hip/hip_runtime.h>
#include <hip/hip_bf16.h>

// ---------------- problem constants ----------------
static constexpr int CB   = 8;     // batch
static constexpr int CN   = 1024;  // atoms per image
static constexpr int CM   = 256;   // neighbors per atom (NTYPES*NN)
static constexpr int CNN  = 128;   // neighbors per type
static constexpr int CD   = 25;    // embedding width
static constexpr int CFH  = 240;   // fitting width
static constexpr float C_RMIN = 5.8f;
static constexpr float C_RMAX = 6.0f;
static constexpr float C_SCALE = 1.0f / 256.0f;     // 4/(NN*NTYPES*4)
static constexpr int CG   = CB * CN;                // 8192 atom rows
static constexpr int CDR  = 416;                    // 400 padded to 13*32

// workspace byte offsets
static constexpr size_t OFF_RI   = 0;                                  // f32 [G][256][4]
static constexpr size_t OFF_RID  = OFF_RI   + (size_t)CG*CM*4*4;       // f32 [G][256][4][3]
static constexpr size_t OFF_ACC  = OFF_RID  + (size_t)CG*CM*12*4;      // f32 [G][4][32]
static constexpr size_t OFF_DRT  = OFF_ACC  + (size_t)CG*128*4;        // f16 [G][416]
static constexpr size_t OFF_H0   = OFF_DRT  + (size_t)CG*CDR*2;        // f32 [G][240]
static constexpr size_t OFF_H1   = OFF_H0   + (size_t)CG*CFH*4;
static constexpr size_t OFF_H2   = OFF_H1   + (size_t)CG*CFH*4;
static constexpr size_t OFF_DDRT = OFF_H2   + (size_t)CG*CFH*4;        // f32 [G][400]
static constexpr size_t OFF_DACC = OFF_DDRT + (size_t)CG*400*4;        // f32 [G][4][32]
static constexpr size_t OFF_DE   = OFF_DACC + (size_t)CG*128*4;        // f32 [G][256][4]

// ---------------- WMMA helpers (gfx1250, wave32) ----------------
typedef _Float16 v16h __attribute__((ext_vector_type(16)));
typedef _Float16 v8h  __attribute__((ext_vector_type(8)));
typedef float    v8f  __attribute__((ext_vector_type(8)));

#define DEV __device__ __forceinline__

DEV v8f v8f_zero() { v8f z = {0.f,0.f,0.f,0.f,0.f,0.f,0.f,0.f}; return z; }

// Branch-free tanh on the hardware transcendental pipe:
// tanh(x) = 1 - 2 / (exp(2x)+1) = 1 - 2*rcp(exp2(x*2*log2(e)) + 1)
// Saturates cleanly: x>>0 -> exp2=inf -> rcp=0 -> 1 ; x<<0 -> exp2=0 -> -1.
DEV float fast_tanh(float x) {
  float e = __builtin_amdgcn_exp2f(x * 2.885390082f);   // 2*log2(e)
  return 1.f - 2.f * __builtin_amdgcn_rcpf(e + 1.f);
}

DEV float fast_rcp(float x) { return __builtin_amdgcn_rcpf(x); }

// A fragment: 16x32 f16, A stored row-major in LDS with row stride lda (halves).
// Contiguous k order used here (same as B): lane<16 -> row=lane, K 0..15 ; lane>=16 -> row=lane-16, K 16..31
DEV v16h load_a_frag(const _Float16* A, int lda, int mBase, int kBase) {
  const int lane = threadIdx.x & 31;
  const int row  = mBase + (lane & 15);
  const int klo  = (lane & 16) ? 16 : 0;
  const _Float16* p = A + (size_t)row * lda + kBase + klo;
  v8h lo = *(const v8h*)p;
  v8h hi = *(const v8h*)(p + 8);
  return __builtin_shufflevector(lo, hi, 0,1,2,3,4,5,6,7,8,9,10,11,12,13,14,15);
}

// B fragment: 32x16 f16. Bt is B transposed (Bt[n][k]) row-major, stride ldb halves.
// lane<16 -> n=lane, K 0..15 ; lane>=16 -> n=lane-16, K 16..31 : contiguous 32B per lane.
DEV v16h load_b_frag(const _Float16* Bt, int ldb, int nBase, int kBase) {
  const int lane = threadIdx.x & 31;
  const int n    = nBase + (lane & 15);
  const int koff = (lane & 16) ? 16 : 0;
  return *(const v16h*)(Bt + (size_t)n * ldb + kBase + koff);
}

DEV v8f wmma_f16(v16h a, v16h b, v8f c) {
  return __builtin_amdgcn_wmma_f32_16x16x32_f16(false, a, false, b, (short)0, c, false, false);
}

// ---------------- K0: zero output ----------------
__global__ void k0_zero(float* p, int n) {
  int i = blockIdx.x * blockDim.x + threadIdx.x;
  if (i < n) p[i] = 0.f;
}

// ---------------- K1: geometry (Ri, Ri_d) ----------------
__global__ void __launch_bounds__(256)
k1_geom(const float* __restrict__ image_dR, const int* __restrict__ list_neigh,
        const float* __restrict__ davg, const float* __restrict__ dstd,
        float* __restrict__ Ri, float* __restrict__ Rid)
{
  size_t idx = (size_t)blockIdx.x * 256 + threadIdx.x;   // < B*N*M
  const int m = (int)(idx & 255);
  const int n = (int)((idx >> 8) & (CN - 1));
  const int t = (n >= 512) ? 1 : 0;

  const float* dr = image_dR + idx * 3;
  float d0 = dr[0], d1 = dr[1], d2 = dr[2];
  int ng = list_neigh[idx];
  float mf = (ng > 0) ? 1.f : 0.f;

  float dR2  = d0*d0 + d1*d1 + d2*d2;
  float safe = (ng > 0) ? dR2 : 1.f;
  float Rij  = sqrtf(safe);
  float inr  = mf * fast_rcp(Rij);
  float x    = Rij * mf;
  float rs   = fast_rcp(safe) * mf;
  float rx0  = d0 * rs, rx1 = d1 * rs, rx2 = d2 * rs;
  float inr2 = inr*inr, inr4 = inr2*inr2, inr3 = inr4*x;

  float u = (x - C_RMIN) * (1.f / (C_RMAX - C_RMIN));
  float poly  = u*u*u*(-6.f*u*u + 15.f*u - 10.f) + 1.f;
  float dpoly = (3.f*u*u*(-6.f*u*u + 15.f*u - 10.f) + u*u*u*(-12.f*u + 15.f)) * (1.f / (C_RMAX - C_RMIN));
  bool mid = (x >= C_RMIN) && (x < C_RMAX);
  float vv  = ((x < C_RMIN) ? 1.f : (mid ? poly : 0.f)) * mf;
  float dvv = (mid ? dpoly : 0.f) * mf;

  float pre[4] = { inr, rx0, rx1, rx2 };
  float dv[3]  = { d0, d1, d2 };
  float cm[3]  = { dvv*inr*d0, dvv*inr*d1, dvv*inr*d2 };

  float rid[4][3];
  #pragma unroll
  for (int j = 0; j < 3; ++j) rid[0][j] = dv[j]*(inr3*vv) - pre[0]*cm[j];
  #pragma unroll
  for (int i = 0; i < 3; ++i)
    #pragma unroll
    for (int j = 0; j < 3; ++j) {
      float outer = 2.f*dv[i]*dv[j]*inr4 - ((i == j) ? inr2 : 0.f);
      rid[1+i][j] = outer*vv - pre[1+i]*cm[j];
    }

  const int si = (t*CM + m)*4;
  #pragma unroll
  for (int d = 0; d < 4; ++d) {
    float rstd = fast_rcp(dstd[si + d]);
    Ri[idx*4 + d] = (pre[d]*vv - davg[si + d]) * rstd;
    #pragma unroll
    for (int j = 0; j < 3; ++j) Rid[idx*12 + d*3 + j] = rid[d][j] * mf * rstd;
  }
}

// ---------------- K2: embedding forward + acc + DRt ----------------
__global__ void __launch_bounds__(128)
k2_embed_fwd(const float* __restrict__ Ri,
             const float* __restrict__ eW0, const float* __restrict__ eb0,
             const float* __restrict__ eW1, const float* __restrict__ eb1,
             const float* __restrict__ eW2, const float* __restrict__ eb2,
             float* __restrict__ accws, _Float16* __restrict__ drt)
{
  __shared__ __align__(32) _Float16 h0buf[128][32];
  __shared__ __align__(32) _Float16 h1buf[128][32];
  __shared__ __align__(32) _Float16 Gt[32][128];
  __shared__ __align__(32) _Float16 blkT[16][128];
  __shared__ __align__(32) _Float16 Wt1[32][32];
  __shared__ __align__(32) _Float16 Wt2[32][32];
  __shared__ float b0L[32], b1L[32], b2L[32], W0L[32];
  __shared__ float accL[4][32];

  const int g = blockIdx.x;
  const int n = g & (CN - 1);
  const int t = (n >= 512) ? 1 : 0;
  const int tid  = threadIdx.x;
  const int wv   = tid >> 5;
  const int lane = tid & 31;

  v8f cacc = v8f_zero();
  for (int i = tid; i < 12*128; i += 128) blkT[4 + (i >> 7)][i & 127] = (_Float16)0.f;

  for (int t1 = 0; t1 < 2; ++t1) {
    const int p = t*2 + t1;
    __syncthreads();
    for (int i = tid; i < 1024; i += 128) {
      int r = i >> 5, c = i & 31;
      _Float16 w1 = (_Float16)0.f, w2 = (_Float16)0.f;
      if (r < 25 && c < 25) {
        w1 = (_Float16)eW1[((size_t)p*CD + c)*CD + r];   // Wt[n][k] = W[k][n]
        w2 = (_Float16)eW2[((size_t)p*CD + c)*CD + r];
      }
      Wt1[r][c] = w1; Wt2[r][c] = w2;
    }
    if (tid < 32) {
      b0L[tid] = (tid < 25) ? eb0[p*CD + tid] : 0.f;
      b1L[tid] = (tid < 25) ? eb1[p*CD + tid] : 0.f;
      b2L[tid] = (tid < 25) ? eb2[p*CD + tid] : 0.f;
      W0L[tid] = (tid < 25) ? eW0[p*CD + tid] : 0.f;
    }
    const float* rp = Ri + ((size_t)g*CM + (size_t)t1*CNN + tid)*4;
    float bl0 = rp[0], bl1 = rp[1], bl2 = rp[2], bl3 = rp[3];
    blkT[0][tid] = (_Float16)bl0; blkT[1][tid] = (_Float16)bl1;
    blkT[2][tid] = (_Float16)bl2; blkT[3][tid] = (_Float16)bl3;
    __syncthreads();
    #pragma unroll
    for (int e = 0; e < 32; ++e) {
      float h = (e < 25) ? fast_tanh(bl0*W0L[e] + b0L[e]) : 0.f;
      h0buf[tid][e] = (_Float16)h;
    }
    __syncthreads();
    // h1 = h0 + tanh(h0 @ W1 + b1)
    #pragma unroll
    for (int mi = 0; mi < 2; ++mi) {
      const int mt = wv + mi*4;
      #pragma unroll
      for (int nt = 0; nt < 2; ++nt) {
        v16h a  = load_a_frag(&h0buf[0][0], 32, mt*16, 0);
        v16h bb = load_b_frag(&Wt1[0][0], 32, nt*16, 0);
        v8f c = v8f_zero(); c = wmma_f16(a, bb, c);
        #pragma unroll
        for (int i = 0; i < 8; ++i) {
          int row = mt*16 + i + ((lane & 16) ? 8 : 0);
          int col = nt*16 + (lane & 15);
          float h0v = (float)h0buf[row][col];
          h1buf[row][col] = (_Float16)(h0v + fast_tanh(c[i] + b1L[col]));
        }
      }
    }
    __syncthreads();
    // G = h1 + tanh(h1 @ W2 + b2)  -> stored transposed (Gt) for acc B-frags
    #pragma unroll
    for (int mi = 0; mi < 2; ++mi) {
      const int mt = wv + mi*4;
      #pragma unroll
      for (int nt = 0; nt < 2; ++nt) {
        v16h a  = load_a_frag(&h1buf[0][0], 32, mt*16, 0);
        v16h bb = load_b_frag(&Wt2[0][0], 32, nt*16, 0);
        v8f c = v8f_zero(); c = wmma_f16(a, bb, c);
        #pragma unroll
        for (int i = 0; i < 8; ++i) {
          int row = mt*16 + i + ((lane & 16) ? 8 : 0);
          int col = nt*16 + (lane & 15);
          float h1v = (float)h1buf[row][col];
          Gt[col][row] = (_Float16)(h1v + fast_tanh(c[i] + b2L[col]));
        }
      }
    }
    __syncthreads();
    // acc += blk^T @ G  (M=16 pad of 4, N=32, K=128)
    if (wv < 2) {
      #pragma unroll
      for (int kt = 0; kt < 4; ++kt) {
        v16h a  = load_a_frag(&blkT[0][0], 128, 0, kt*32);
        v16h bb = load_b_frag(&Gt[0][0], 128, wv*16, kt*32);
        cacc = wmma_f16(a, bb, cacc);
      }
    }
    __syncthreads();
  }
  if (wv < 2 && lane < 16) {
    const int col = wv*16 + lane;
    #pragma unroll
    for (int i = 0; i < 4; ++i) {
      float v = cacc[i] * C_SCALE;
      accL[i][col] = v;
      accws[((size_t)g*4 + i)*32 + col] = v;
    }
  }
  __syncthreads();
  for (int idx = tid; idx < CDR; idx += 128) {
    float v = 0.f;
    if (idx < 400) {
      int e = idx >> 4, f = idx & 15;
      v = accL[0][e]*accL[0][f] + accL[1][e]*accL[1][f]
        + accL[2][e]*accL[2][f] + accL[3][e]*accL[3][f];
    }
    drt[(size_t)g*CDR + idx] = (_Float16)v;
  }
}

// ---------------- K3: fitting forward (fused 3 layers + Ei) ----------------
__global__ void __launch_bounds__(256)
k3_fit_fwd(const _Float16* __restrict__ drt,
           const float* __restrict__ fW0, const float* __restrict__ fb0,
           const float* __restrict__ fW1, const float* __restrict__ fb1,
           const float* __restrict__ fW2, const float* __restrict__ fb2,
           const float* __restrict__ fW3, const float* __restrict__ fb3,
           const float* __restrict__ ener_shift,
           float* __restrict__ h0w, float* __restrict__ h1w, float* __restrict__ h2w,
           float* __restrict__ out)
{
  __shared__ __align__(32) _Float16 hbuf[64][256];
  __shared__ __align__(32) _Float16 Aslab[64][32];
  __shared__ __align__(32) _Float16 Wslab[240][32];
  __shared__ float bL0[240], bL1[240], bL2[240], fW3L[240];

  const int g0 = blockIdx.x * 64;
  const int t  = ((g0 & (CN - 1)) >= 512) ? 1 : 0;
  const int tid  = threadIdx.x;
  const int wv   = tid >> 5;
  const int lane = tid & 31;
  const int mt   = wv >> 1;
  const int n0   = (wv & 1) ? 8 : 0;
  const int ncnt = (wv & 1) ? 7 : 8;

  for (int i = tid; i < 240; i += 256) {
    bL0[i] = fb0[t*CFH + i]; bL1[i] = fb1[t*CFH + i];
    bL2[i] = fb2[t*CFH + i]; fW3L[i] = fW3[t*CFH + i];
  }
  for (int i = tid; i < 64*16; i += 256) hbuf[i >> 4][240 + (i & 15)] = (_Float16)0.f;

  v8f c[8];
  // layer 0: DRt @ fW0
  #pragma unroll
  for (int j = 0; j < 8; ++j) c[j] = v8f_zero();
  for (int kt = 0; kt < 13; ++kt) {
    __syncthreads();
    for (int i = tid; i < 64*32; i += 256) {
      int r = i >> 5, kk = i & 31;
      Aslab[r][kk] = drt[(size_t)(g0 + r)*CDR + kt*32 + kk];
    }
    for (int i = tid; i < 240*32; i += 256) {
      int nn = i >> 5, kk = i & 31, k = kt*32 + kk;
      Wslab[nn][kk] = (k < 400) ? (_Float16)fW0[((size_t)t*400 + k)*CFH + nn] : (_Float16)0.f;
    }
    __syncthreads();
    v16h a = load_a_frag(&Aslab[0][0], 32, mt*16, 0);
    #pragma unroll
    for (int j = 0; j < 8; ++j) if (j < ncnt) {
      v16h bb = load_b_frag(&Wslab[0][0], 32, (n0 + j)*16, 0);
      c[j] = wmma_f16(a, bb, c[j]);
    }
  }
  __syncthreads();
  #pragma unroll
  for (int j = 0; j < 8; ++j) if (j < ncnt)
    #pragma unroll
    for (int i = 0; i < 8; ++i) {
      int row = mt*16 + i + ((lane & 16) ? 8 : 0);
      int col = (n0 + j)*16 + (lane & 15);
      float v = fast_tanh(c[j][i] + bL0[col]);
      hbuf[row][col] = (_Float16)v;
      h0w[(size_t)(g0 + row)*CFH + col] = v;
    }
  __syncthreads();
  // layers 1 and 2 (residual)
  for (int layer = 1; layer <= 2; ++layer) {
    const float* W  = (layer == 1) ? fW1 : fW2;
    const float* bl = (layer == 1) ? bL1 : bL2;
    float* hw       = (layer == 1) ? h1w : h2w;
    #pragma unroll
    for (int j = 0; j < 8; ++j) c[j] = v8f_zero();
    for (int kt = 0; kt < 8; ++kt) {
      __syncthreads();
      for (int i = tid; i < 240*32; i += 256) {
        int nn = i >> 5, kk = i & 31, k = kt*32 + kk;
        Wslab[nn][kk] = (k < 240) ? (_Float16)W[((size_t)t*CFH + k)*CFH + nn] : (_Float16)0.f;
      }
      __syncthreads();
      v16h a = load_a_frag(&hbuf[0][0], 256, mt*16, kt*32);
      #pragma unroll
      for (int j = 0; j < 8; ++j) if (j < ncnt) {
        v16h bb = load_b_frag(&Wslab[0][0], 32, (n0 + j)*16, 0);
        c[j] = wmma_f16(a, bb, c[j]);
      }
    }
    __syncthreads();
    #pragma unroll
    for (int j = 0; j < 8; ++j) if (j < ncnt)
      #pragma unroll
      for (int i = 0; i < 8; ++i) {
        int row = mt*16 + i + ((lane & 16) ? 8 : 0);
        int col = (n0 + j)*16 + (lane & 15);
        float prev = (float)hbuf[row][col];
        float v = prev + fast_tanh(c[j][i] + bl[col]);
        hbuf[row][col] = (_Float16)v;
        hw[(size_t)(g0 + row)*CFH + col] = v;
      }
    __syncthreads();
  }
  // Ei = h2 @ fW3 + fb3 + shift ; Etot via atomic
  if (tid < 64) {
    int g = g0 + tid;
    float s = fb3[t] + ener_shift[t];
    for (int j = 0; j < 240; ++j) s += (float)hbuf[tid][j] * fW3L[j];
    out[8 + g] = s;
    atomicAdd(&out[g >> 10], s);
  }
}

// ---------------- K4: fitting backward -> dDRt ----------------
__global__ void __launch_bounds__(128)
k4_fit_bwd(const float* __restrict__ h0w, const float* __restrict__ h1w, const float* __restrict__ h2w,
           const float* __restrict__ fW0, const float* __restrict__ fW1, const float* __restrict__ fW2,
           const float* __restrict__ fW3, float* __restrict__ ddrt)
{
  __shared__ __align__(32) _Float16 ubuf[32][256];
  __shared__ __align__(32) _Float16 dbuf[32][256];
  __shared__ __align__(32) _Float16 Wslab[400][32];
  __shared__ float fW3L[240];

  const int g0 = blockIdx.x * 32;
  const int t  = ((g0 & (CN - 1)) >= 512) ? 1 : 0;
  const int tid  = threadIdx.x;
  const int wv   = tid >> 5;
  const int lane = tid & 31;
  const int mt   = wv >> 1;
  const int n0   = (wv & 1) ? 8 : 0;
  const int ncnt = (wv & 1) ? 7 : 8;

  for (int i = tid; i < 240; i += 128) fW3L[i] = fW3[t*CFH + i];
  for (int i = tid; i < 32*16; i += 128) {
    ubuf[i >> 4][240 + (i & 15)] = (_Float16)0.f;
    dbuf[i >> 4][240 + (i & 15)] = (_Float16)0.f;
  }
  __syncthreads();
  // u2 = dh2 * (1 - t2^2), dh2 = fW3 (broadcast), t2 = h2 - h1
  for (int i = tid; i < 32*240; i += 128) {
    int r = i / 240, cc = i % 240;
    size_t g = (size_t)(g0 + r);
    float t2 = h2w[g*CFH + cc] - h1w[g*CFH + cc];
    ubuf[r][cc] = (_Float16)(fW3L[cc] * (1.f - t2*t2));
  }

  v8f c[13];
  // v2 = u2 @ fW2^T  (Bt layout == fW2 natural row-major)
  #pragma unroll
  for (int j = 0; j < 13; ++j) c[j] = v8f_zero();
  for (int kt = 0; kt < 8; ++kt) {
    __syncthreads();
    for (int i = tid; i < 240*32; i += 128) {
      int nn = i >> 5, kk = i & 31, k = kt*32 + kk;
      Wslab[nn][kk] = (k < 240) ? (_Float16)fW2[((size_t)t*CFH + nn)*CFH + k] : (_Float16)0.f;
    }
    __syncthreads();
    v16h a = load_a_frag(&ubuf[0][0], 256, mt*16, kt*32);
    #pragma unroll
    for (int j = 0; j < 8; ++j) if (j < ncnt) {
      v16h bb = load_b_frag(&Wslab[0][0], 32, (n0 + j)*16, 0);
      c[j] = wmma_f16(a, bb, c[j]);
    }
  }
  __syncthreads();
  #pragma unroll
  for (int j = 0; j < 8; ++j) if (j < ncnt)
    #pragma unroll
    for (int i = 0; i < 8; ++i) {
      int row = mt*16 + i + ((lane & 16) ? 8 : 0);
      int col = (n0 + j)*16 + (lane & 15);
      size_t g = (size_t)(g0 + row);
      float dh1 = fW3L[col] + c[j][i];
      dbuf[row][col] = (_Float16)dh1;
      float t1a = h1w[g*CFH + col] - h0w[g*CFH + col];
      ubuf[row][col] = (_Float16)(dh1 * (1.f - t1a*t1a));   // u1
    }
  __syncthreads();
  // v1 = u1 @ fW1^T -> u0
  #pragma unroll
  for (int j = 0; j < 13; ++j) c[j] = v8f_zero();
  for (int kt = 0; kt < 8; ++kt) {
    __syncthreads();
    for (int i = tid; i < 240*32; i += 128) {
      int nn = i >> 5, kk = i & 31, k = kt*32 + kk;
      Wslab[nn][kk] = (k < 240) ? (_Float16)fW1[((size_t)t*CFH + nn)*CFH + k] : (_Float16)0.f;
    }
    __syncthreads();
    v16h a = load_a_frag(&ubuf[0][0], 256, mt*16, kt*32);
    #pragma unroll
    for (int j = 0; j < 8; ++j) if (j < ncnt) {
      v16h bb = load_b_frag(&Wslab[0][0], 32, (n0 + j)*16, 0);
      c[j] = wmma_f16(a, bb, c[j]);
    }
  }
  __syncthreads();
  #pragma unroll
  for (int j = 0; j < 8; ++j) if (j < ncnt)
    #pragma unroll
    for (int i = 0; i < 8; ++i) {
      int row = mt*16 + i + ((lane & 16) ? 8 : 0);
      int col = (n0 + j)*16 + (lane & 15);
      size_t g = (size_t)(g0 + row);
      float dh0 = (float)dbuf[row][col] + c[j][i];
      float h0v = h0w[g*CFH + col];
      ubuf[row][col] = (_Float16)(dh0 * (1.f - h0v*h0v));   // u0
    }
  __syncthreads();
  // dDRt = u0 @ fW0^T  (N = 400 = 25 exact tiles)
  const int n0c = (wv & 1) ? 13 : 0;
  const int ncc = (wv & 1) ? 12 : 13;
  #pragma unroll
  for (int j = 0; j < 13; ++j) c[j] = v8f_zero();
  for (int kt = 0; kt < 8; ++kt) {
    __syncthreads();
    for (int i = tid; i < 400*32; i += 128) {
      int nn = i >> 5, kk = i & 31, k = kt*32 + kk;
      Wslab[nn][kk] = (k < 240) ? (_Float16)fW0[((size_t)t*400 + nn)*CFH + k] : (_Float16)0.f;
    }
    __syncthreads();
    v16h a = load_a_frag(&ubuf[0][0], 256, mt*16, kt*32);
    #pragma unroll
    for (int j = 0; j < 13; ++j) if (j < ncc) {
      v16h bb = load_b_frag(&Wslab[0][0], 32, (n0c + j)*16, 0);
      c[j] = wmma_f16(a, bb, c[j]);
    }
  }
  #pragma unroll
  for (int j = 0; j < 13; ++j) if (j < ncc)
    #pragma unroll
    for (int i = 0; i < 8; ++i) {
      int row = mt*16 + i + ((lane & 16) ? 8 : 0);
      int col = (n0c + j)*16 + (lane & 15);
      ddrt[(size_t)(g0 + row)*400 + col] = c[j][i];
    }
}

// ---------------- K5: dacc from dDRt and acc ----------------
__global__ void __launch_bounds__(128)
k5_dacc(const float* __restrict__ accws, const float* __restrict__ ddrt,
        float* __restrict__ daccws)
{
  __shared__ float accL[4][32];
  __shared__ float dd[400];
  const int g = blockIdx.x;
  const int tid = threadIdx.x;
  accL[tid >> 5][tid & 31] = accws[(size_t)g*128 + tid];
  for (int i = tid; i < 400; i += 128) dd[i] = ddrt[(size_t)g*400 + i];
  __syncthreads();
  const int d = tid >> 5, e = tid & 31;
  float v = 0.f;
  if (e < 25) {
    #pragma unroll
    for (int f = 0; f < 16; ++f) v += dd[e*16 + f] * accL[d][f];
    if (e < 16) {
      #pragma unroll
      for (int e2 = 0; e2 < 25; ++e2) v += dd[e2*16 + e] * accL[d][e2];
    }
  }
  daccws[(size_t)g*128 + tid] = v * C_SCALE;   // pre-scale grad back to raw acc
}

// ---------------- K6: embedding backward (recompute fwd) -> dE ----------------
__global__ void __launch_bounds__(128)
k6_embed_bwd(const float* __restrict__ Ri, const float* __restrict__ daccws,
             const float* __restrict__ eW0, const float* __restrict__ eb0,
             const float* __restrict__ eW1, const float* __restrict__ eb1,
             const float* __restrict__ eW2, const float* __restrict__ eb2,
             float* __restrict__ dE)
{
  __shared__ __align__(32) _Float16 h0buf[128][32];
  __shared__ __align__(32) _Float16 h1buf[128][32];
  __shared__ __align__(32) _Float16 t2buf[128][32];
  __shared__ __align__(32) _Float16 Gbuf[128][32];
  __shared__ __align__(32) _Float16 ubuf[128][32];
  __shared__ __align__(32) _Float16 d1buf[128][32];
  __shared__ __align__(32) _Float16 blkT[16][128];
  __shared__ __align__(32) _Float16 Wt1[32][32], Wt2[32][32], Wb1[32][32], Wb2[32][32];
  __shared__ float b0L[32], b1L[32], b2L[32], W0L[32];
  __shared__ float daccL[4][32];

  const int g = blockIdx.x;
  const int n = g & (CN - 1);
  const int t = (n >= 512) ? 1 : 0;
  const int tid  = threadIdx.x;
  const int wv   = tid >> 5;
  const int lane = tid & 31;

  daccL[tid >> 5][tid & 31] = daccws[(size_t)g*128 + tid];
  for (int i = tid; i < 12*128; i += 128) blkT[4 + (i >> 7)][i & 127] = (_Float16)0.f;

  for (int t1 = 0; t1 < 2; ++t1) {
    const int p = t*2 + t1;
    __syncthreads();
    for (int i = tid; i < 1024; i += 128) {
      int r = i >> 5, cc = i & 31;
      _Float16 f1 = (_Float16)0.f, f2 = (_Float16)0.f, g1 = (_Float16)0.f, g2 = (_Float16)0.f;
      if (r < 25 && cc < 25) {
        f1 = (_Float16)eW1[((size_t)p*CD + cc)*CD + r];   // forward: W^T layout
        f2 = (_Float16)eW2[((size_t)p*CD + cc)*CD + r];
        g1 = (_Float16)eW1[((size_t)p*CD + r)*CD + cc];   // backward (@W^T): natural
        g2 = (_Float16)eW2[((size_t)p*CD + r)*CD + cc];
      }
      Wt1[r][cc] = f1; Wt2[r][cc] = f2; Wb1[r][cc] = g1; Wb2[r][cc] = g2;
    }
    if (tid < 32) {
      b0L[tid] = (tid < 25) ? eb0[p*CD + tid] : 0.f;
      b1L[tid] = (tid < 25) ? eb1[p*CD + tid] : 0.f;
      b2L[tid] = (tid < 25) ? eb2[p*CD + tid] : 0.f;
      W0L[tid] = (tid < 25) ? eW0[p*CD + tid] : 0.f;
    }
    const float* rp = Ri + ((size_t)g*CM + (size_t)t1*CNN + tid)*4;
    float bl0 = rp[0], bl1 = rp[1], bl2 = rp[2], bl3 = rp[3];
    blkT[0][tid] = (_Float16)bl0; blkT[1][tid] = (_Float16)bl1;
    blkT[2][tid] = (_Float16)bl2; blkT[3][tid] = (_Float16)bl3;
    __syncthreads();
    #pragma unroll
    for (int e = 0; e < 32; ++e) {
      float h = (e < 25) ? fast_tanh(bl0*W0L[e] + b0L[e]) : 0.f;
      h0buf[tid][e] = (_Float16)h;
    }
    __syncthreads();
    // recompute h1
    #pragma unroll
    for (int mi = 0; mi < 2; ++mi) {
      const int mt = wv + mi*4;
      #pragma unroll
      for (int nt = 0; nt < 2; ++nt) {
        v16h a  = load_a_frag(&h0buf[0][0], 32, mt*16, 0);
        v16h bb = load_b_frag(&Wt1[0][0], 32, nt*16, 0);
        v8f c = v8f_zero(); c = wmma_f16(a, bb, c);
        #pragma unroll
        for (int i = 0; i < 8; ++i) {
          int row = mt*16 + i + ((lane & 16) ? 8 : 0);
          int col = nt*16 + (lane & 15);
          float h0v = (float)h0buf[row][col];
          h1buf[row][col] = (_Float16)(h0v + fast_tanh(c[i] + b1L[col]));
        }
      }
    }
    __syncthreads();
    // recompute t2 and G
    #pragma unroll
    for (int mi = 0; mi < 2; ++mi) {
      const int mt = wv + mi*4;
      #pragma unroll
      for (int nt = 0; nt < 2; ++nt) {
        v16h a  = load_a_frag(&h1buf[0][0], 32, mt*16, 0);
        v16h bb = load_b_frag(&Wt2[0][0], 32, nt*16, 0);
        v8f c = v8f_zero(); c = wmma_f16(a, bb, c);
        #pragma unroll
        for (int i = 0; i < 8; ++i) {
          int row = mt*16 + i + ((lane & 16) ? 8 : 0);
          int col = nt*16 + (lane & 15);
          float t2v = fast_tanh(c[i] + b2L[col]);
          t2buf[row][col] = (_Float16)t2v;
          Gbuf[row][col]  = (_Float16)((float)h1buf[row][col] + t2v);
        }
      }
    }
    __syncthreads();
    // per-neighbor: dblk and u2 = dG*(1-t2^2), dG[k][e] = sum_d dacc[d][e]*blk[k][d]
    float dblk0 = 0.f, dblk1 = 0.f, dblk2 = 0.f, dblk3 = 0.f;
    #pragma unroll
    for (int e = 0; e < 25; ++e) {
      float Gv = (float)Gbuf[tid][e];
      dblk0 += daccL[0][e]*Gv; dblk1 += daccL[1][e]*Gv;
      dblk2 += daccL[2][e]*Gv; dblk3 += daccL[3][e]*Gv;
    }
    #pragma unroll
    for (int e = 0; e < 32; ++e) {
      float u = 0.f;
      if (e < 25) {
        float dg = daccL[0][e]*bl0 + daccL[1][e]*bl1 + daccL[2][e]*bl2 + daccL[3][e]*bl3;
        float t2v = (float)t2buf[tid][e];
        u = dg * (1.f - t2v*t2v);
      }
      ubuf[tid][e] = (_Float16)u;
    }
    __syncthreads();
    // v2 = u2 @ W2^T ; dh1 = dG + v2 ; u1 = dh1*(1-t1a^2)
    #pragma unroll
    for (int mi = 0; mi < 2; ++mi) {
      const int mt = wv + mi*4;
      #pragma unroll
      for (int nt = 0; nt < 2; ++nt) {
        v16h a  = load_a_frag(&ubuf[0][0], 32, mt*16, 0);
        v16h bb = load_b_frag(&Wb2[0][0], 32, nt*16, 0);
        v8f c = v8f_zero(); c = wmma_f16(a, bb, c);
        #pragma unroll
        for (int i = 0; i < 8; ++i) {
          int row = mt*16 + i + ((lane & 16) ? 8 : 0);
          int col = nt*16 + (lane & 15);
          float dg = daccL[0][col]*(float)blkT[0][row] + daccL[1][col]*(float)blkT[1][row]
                   + daccL[2][col]*(float)blkT[2][row] + daccL[3][col]*(float)blkT[3][row];
          float dh1 = dg + c[i];
          d1buf[row][col] = (_Float16)dh1;
          float t1a = (float)h1buf[row][col] - (float)h0buf[row][col];
          t2buf[row][col] = (_Float16)(dh1 * (1.f - t1a*t1a));   // u1
        }
      }
    }
    __syncthreads();
    // v1 = u1 @ W1^T ; dh0 = dh1 + v1 ; u0 = dh0*(1-h0^2)
    #pragma unroll
    for (int mi = 0; mi < 2; ++mi) {
      const int mt = wv + mi*4;
      #pragma unroll
      for (int nt = 0; nt < 2; ++nt) {
        v16h a  = load_a_frag(&t2buf[0][0], 32, mt*16, 0);
        v16h bb = load_b_frag(&Wb1[0][0], 32, nt*16, 0);
        v8f c = v8f_zero(); c = wmma_f16(a, bb, c);
        #pragma unroll
        for (int i = 0; i < 8; ++i) {
          int row = mt*16 + i + ((lane & 16) ? 8 : 0);
          int col = nt*16 + (lane & 15);
          float dh0 = (float)d1buf[row][col] + c[i];
          float h0v = (float)h0buf[row][col];
          ubuf[row][col] = (_Float16)(dh0 * (1.f - h0v*h0v));    // u0
        }
      }
    }
    __syncthreads();
    // ds = sum_e u0[e]*W0[e] ; write dE
    float ds = 0.f;
    #pragma unroll
    for (int e = 0; e < 25; ++e) ds += (float)ubuf[tid][e] * W0L[e];
    size_t mo = ((size_t)g*CM + (size_t)t1*CNN + tid)*4;
    dE[mo + 0] = dblk0 + ds;
    dE[mo + 1] = dblk1;
    dE[mo + 2] = dblk2;
    dE[mo + 3] = dblk3;
  }
}

// ---------------- K7: pair forces + scatter ----------------
__global__ void __launch_bounds__(128)
k7_force(const float* __restrict__ dE, const float* __restrict__ Rid,
         const int* __restrict__ neigh, float* __restrict__ out)
{
  __shared__ float fs[3];
  const int g = blockIdx.x;
  const int b = g >> 10;
  const int tid = threadIdx.x;
  if (tid < 3) fs[tid] = 0.f;
  __syncthreads();
  float* F = out + 8 + CG;   // after Etot(8) and Ei(8192)
  for (int m = tid; m < CM; m += 128) {
    size_t mo = (size_t)g*CM + m;
    const float* de = dE + mo*4;
    const float* rd = Rid + mo*12;
    float px = de[0]*rd[0] + de[1]*rd[3] + de[2]*rd[6] + de[3]*rd[9];
    float py = de[0]*rd[1] + de[1]*rd[4] + de[2]*rd[7] + de[3]*rd[10];
    float pz = de[0]*rd[2] + de[1]*rd[5] + de[2]*rd[8] + de[3]*rd[11];
    atomicAdd(&fs[0], -px); atomicAdd(&fs[1], -py); atomicAdd(&fs[2], -pz);
    int j = neigh[mo] - 1;
    if (j >= 0) {
      size_t fo = ((size_t)b*CN + j)*3;
      atomicAdd(&F[fo + 0], px); atomicAdd(&F[fo + 1], py); atomicAdd(&F[fo + 2], pz);
    }
  }
  __syncthreads();
  if (tid < 3) atomicAdd(&F[(size_t)g*3 + tid], fs[tid]);
}

// ---------------- host launcher ----------------
extern "C" void kernel_launch(void* const* d_in, const int* in_sizes, int n_in,
                              void* d_out, int out_size, void* d_ws, size_t ws_size,
                              hipStream_t stream) {
  (void)in_sizes; (void)n_in; (void)out_size; (void)ws_size;
  const float* image_dR  = (const float*)d_in[0];
  const int*   list_neigh= (const int*)  d_in[1];
  const float* davg = (const float*)d_in[2];
  const float* dstd = (const float*)d_in[3];
  const float* eW0 = (const float*)d_in[4];
  const float* eb0 = (const float*)d_in[5];
  const float* eW1 = (const float*)d_in[6];
  const float* eb1 = (const float*)d_in[7];
  const float* eW2 = (const float*)d_in[8];
  const float* eb2 = (const float*)d_in[9];
  const float* fW0 = (const float*)d_in[10];
  const float* fb0 = (const float*)d_in[11];
  const float* fW1 = (const float*)d_in[12];
  const float* fb1 = (const float*)d_in[13];
  const float* fW2 = (const float*)d_in[14];
  const float* fb2 = (const float*)d_in[15];
  const float* fW3 = (const float*)d_in[16];
  const float* fb3 = (const float*)d_in[17];
  const float* ener_shift = (const float*)d_in[18];

  char* ws = (char*)d_ws;
  float*     Ri    = (float*)(ws + OFF_RI);
  float*     Rid   = (float*)(ws + OFF_RID);
  float*     accws = (float*)(ws + OFF_ACC);
  _Float16*  drt   = (_Float16*)(ws + OFF_DRT);
  float*     h0w   = (float*)(ws + OFF_H0);
  float*     h1w   = (float*)(ws + OFF_H1);
  float*     h2w   = (float*)(ws + OFF_H2);
  float*     ddrt  = (float*)(ws + OFF_DDRT);
  float*     daccw = (float*)(ws + OFF_DACC);
  float*     dEw   = (float*)(ws + OFF_DE);
  float*     out   = (float*)d_out;

  const int out_n = 8 + CG + CG*3;   // Etot + Ei + F = 32776
  k0_zero<<<(out_n + 255)/256, 256, 0, stream>>>(out, out_n);
  k1_geom<<<(CG*CM)/256, 256, 0, stream>>>(image_dR, list_neigh, davg, dstd, Ri, Rid);
  k2_embed_fwd<<<CG, 128, 0, stream>>>(Ri, eW0, eb0, eW1, eb1, eW2, eb2, accws, drt);
  k3_fit_fwd<<<CG/64, 256, 0, stream>>>(drt, fW0, fb0, fW1, fb1, fW2, fb2, fW3, fb3,
                                        ener_shift, h0w, h1w, h2w, out);
  k4_fit_bwd<<<CG/32, 128, 0, stream>>>(h0w, h1w, h2w, fW0, fW1, fW2, fW3, ddrt);
  k5_dacc<<<CG, 128, 0, stream>>>(accws, ddrt, daccw);
  k6_embed_bwd<<<CG, 128, 0, stream>>>(Ri, daccw, eW0, eb0, eW1, eb1, eW2, eb2, dEw);
  k7_force<<<CG, 128, 0, stream>>>(dEw, Rid, list_neigh, out);
}
